// PointMambaScan_22892175688209
// MI455X (gfx1250) — compile-verified, matrix operand order checked by reading, not run on hardware
//
#include <hip/hip_runtime.h>
#include <hip/hip_bf16.h>

// ---------------- problem constants ----------------
#define Bq   4
#define Nq   2048
#define DMq  384
#define DIq  768
#define Sq   16
#define Rq   24
#define Kq   4
#define ROWS (Bq * Nq)            // 8192
#define XZW  (2 * DIq)            // 1536
#define XDW  (Rq + 2 * Sq)        // 56
#define XDWP 64                   // padded to WMMA tile width

typedef _Float16 v16h __attribute__((ext_vector_type(16)));
typedef float    v8f  __attribute__((ext_vector_type(8)));

union Frag16 { uint4 q[2]; v16h v; };

// Load a 16x32 f16 WMMA operand fragment (row-major source, ld halfwords).
// Per ISA layout: lanes 0-15 -> row = lane, K = 0..7 (vgpr0-3), 16..23 (vgpr4-7)
//                 lanes 16-31 -> row = lane-16, K = 8..15, 24..31
__device__ __forceinline__ v16h load_frag(const _Float16* __restrict__ base,
                                          int ld, int lane) {
  const int row = lane & 15;
  const int kb  = (lane >> 4) << 3;          // 0 or 8
  const _Float16* p = base + row * ld + kb;
  Frag16 f;
  f.q[0] = *reinterpret_cast<const uint4*>(p);        // K = kb .. kb+7
  f.q[1] = *reinterpret_cast<const uint4*>(p + 16);   // K = kb+16 .. kb+23
  return f.v;
}

__device__ __forceinline__ v8f wmma_f16(v16h a, v16h b, v8f c) {
  return __builtin_amdgcn_wmma_f32_16x16x32_f16(false, a, false, b,
                                                (short)0, c, false, false);
}

// ---------------- 0) weight convert/transpose: W(KxN) -> WT(Npad x K) f16 ----
__global__ void wt_f16_kernel(const float* __restrict__ W, _Float16* __restrict__ WT,
                              int K, int N, int Npad) {
  int id = blockIdx.x * 256 + threadIdx.x;
  if (id >= Npad * K) return;
  int n = id / K, k = id - n * K;
  WT[n * K + k] = (n < N) ? (_Float16)W[k * N + n] : (_Float16)0.0f;
}

// ---------------- 1) morton codes (per batch) ----------------
__global__ void morton_kernel(const float* __restrict__ pos, unsigned* __restrict__ codes) {
  const int b = blockIdx.x;
  __shared__ int smn[3];
  if (threadIdx.x == 0) { smn[0] = 0x7fffffff; smn[1] = 0x7fffffff; smn[2] = 0x7fffffff; }
  __syncthreads();
  int lm0 = 0x7fffffff, lm1 = 0x7fffffff, lm2 = 0x7fffffff;
  for (int i = threadIdx.x; i < Nq; i += 256) {
    const float* p = pos + (size_t)(b * Nq + i) * 3;
    int g0 = (int)floorf(p[0] * 50.0f);   // 1/GRID = 50
    int g1 = (int)floorf(p[1] * 50.0f);
    int g2 = (int)floorf(p[2] * 50.0f);
    lm0 = min(lm0, g0); lm1 = min(lm1, g1); lm2 = min(lm2, g2);
  }
  atomicMin(&smn[0], lm0); atomicMin(&smn[1], lm1); atomicMin(&smn[2], lm2);
  __syncthreads();
  int m0 = smn[0], m1 = smn[1], m2 = smn[2];
  for (int i = threadIdx.x; i < Nq; i += 256) {
    const float* p = pos + (size_t)(b * Nq + i) * 3;
    int g0 = min(max((int)floorf(p[0] * 50.0f) - m0, 0), 255);
    int g1 = min(max((int)floorf(p[1] * 50.0f) - m1, 0), 255);
    int g2 = min(max((int)floorf(p[2] * 50.0f) - m2, 0), 255);
    unsigned c = 0;
#pragma unroll
    for (int bit = 0; bit < 8; ++bit) {
      c |= ((unsigned)((g0 >> bit) & 1)) << (3 * bit);
      c |= ((unsigned)((g1 >> bit) & 1)) << (3 * bit + 1);
      c |= ((unsigned)((g2 >> bit) & 1)) << (3 * bit + 2);
    }
    codes[b * Nq + i] = c;
  }
}

// ---------------- 2) per-batch bitonic sort in LDS ----------------
__global__ __launch_bounds__(1024)
void sort_kernel(const unsigned* __restrict__ codes, int* __restrict__ order,
                 int* __restrict__ inv) {
  const int b = blockIdx.x;
  __shared__ unsigned long long key[Nq];
  for (int i = threadIdx.x; i < Nq; i += 1024)
    key[i] = ((unsigned long long)codes[b * Nq + i] << 32) | (unsigned)i;
  __syncthreads();
  for (int k = 2; k <= Nq; k <<= 1) {
    for (int j = k >> 1; j > 0; j >>= 1) {
      for (int t = threadIdx.x; t < Nq; t += 1024) {
        int ixj = t ^ j;
        if (ixj > t) {
          unsigned long long a = key[t], c = key[ixj];
          bool up = ((t & k) == 0);
          if (up ? (a > c) : (a < c)) { key[t] = c; key[ixj] = a; }
        }
      }
      __syncthreads();
    }
  }
  for (int i = threadIdx.x; i < Nq; i += 1024) {
    int idx = (int)(key[i] & 0xffffffffull);
    order[b * Nq + i] = idx;
    inv[b * Nq + idx] = i;
  }
}

// ---------------- 3) gather + layernorm ----------------
__global__ void gather_ln_kernel(const float* __restrict__ feat, const int* __restrict__ order,
                                 const float* __restrict__ g, const float* __restrict__ bt,
                                 float* __restrict__ feat_s, _Float16* __restrict__ h16) {
  const int row = blockIdx.x;            // 0..ROWS-1
  const int b   = row >> 11;
  const int src = order[row];
  const float* fin = feat + (size_t)(b * Nq + src) * DMq;
  const int tid = threadIdx.x;           // 128 threads, 3 elems each
  __shared__ float red[128];
  float v0 = fin[tid], v1 = fin[tid + 128], v2 = fin[tid + 256];
  red[tid] = v0 + v1 + v2;
  __syncthreads();
  for (int o = 64; o > 0; o >>= 1) { if (tid < o) red[tid] += red[tid + o]; __syncthreads(); }
  const float mean = red[0] * (1.0f / DMq);
  __syncthreads();
  float d0 = v0 - mean, d1 = v1 - mean, d2 = v2 - mean;
  red[tid] = d0 * d0 + d1 * d1 + d2 * d2;
  __syncthreads();
  for (int o = 64; o > 0; o >>= 1) { if (tid < o) red[tid] += red[tid + o]; __syncthreads(); }
  const float rstd = rsqrtf(red[0] * (1.0f / DMq) + 1e-5f);
  float* fs = feat_s + (size_t)row * DMq;
  _Float16* hh = h16 + (size_t)row * DMq;
  fs[tid]       = v0; hh[tid]       = (_Float16)(d0 * rstd * g[tid]       + bt[tid]);
  fs[tid + 128] = v1; hh[tid + 128] = (_Float16)(d1 * rstd * g[tid + 128] + bt[tid + 128]);
  fs[tid + 256] = v2; hh[tid + 256] = (_Float16)(d2 * rstd * g[tid + 256] + bt[tid + 256]);
}

// ---------------- 4) GEMM xz = h(8192x384) * W_in -> (8192x1536) ----------------
__global__ void gemm_xz_kernel(const _Float16* __restrict__ H, const _Float16* __restrict__ WT,
                               float* __restrict__ XZ) {
  const int lane = threadIdx.x & 31, wave = threadIdx.x >> 5;
  const int row0 = blockIdx.x * 16;
  const int col0 = (blockIdx.y * 8 + wave) * 64;
  v8f acc[4] = {};
  for (int k = 0; k < DMq; k += 32) {
    v16h a = load_frag(H + (size_t)row0 * DMq + k, DMq, lane);
#pragma unroll
    for (int t = 0; t < 4; ++t) {
      v16h bf = load_frag(WT + (size_t)(col0 + 16 * t) * DMq + k, DMq, lane);
      acc[t] = wmma_f16(a, bf, acc[t]);
    }
  }
  const int n = lane & 15, half = lane >> 4;
#pragma unroll
  for (int t = 0; t < 4; ++t)
#pragma unroll
    for (int r = 0; r < 8; ++r)
      XZ[(size_t)(row0 + r + 8 * half) * XZW + col0 + 16 * t + n] = acc[t][r];
}

// ---------------- 5) causal conv (K=4) + SiLU ----------------
__global__ void conv_silu_kernel(const float* __restrict__ XZ, const float* __restrict__ cw,
                                 const float* __restrict__ cb,
                                 float* __restrict__ xc32, _Float16* __restrict__ xc16) {
  int id = blockIdx.x * 256 + threadIdx.x;
  if (id >= ROWS * DIq) return;
  int d = id % DIq, row = id / DIq, t = row & (Nq - 1), b = row >> 11;
  float acc = cb[d];
#pragma unroll
  for (int k = 0; k < Kq; ++k) {
    int tt = t - (Kq - 1) + k;
    if (tt >= 0) acc += XZ[(size_t)((b << 11) + tt) * XZW + d] * cw[d * Kq + k];
  }
  float s = acc / (1.0f + __expf(-acc));
  xc32[id] = s;
  xc16[id] = (_Float16)s;
}

// ---------------- 6) GEMM xdbl = xc(8192x768) * W_x -> (8192x56) ----------------
__global__ void gemm_xdbl_kernel(const _Float16* __restrict__ XC, const _Float16* __restrict__ WT,
                                 float* __restrict__ XD) {
  const int lane = threadIdx.x & 31, wave = threadIdx.x >> 5;
  const int row0 = (blockIdx.x * 8 + wave) * 16;
  v8f acc[4] = {};
  for (int k = 0; k < DIq; k += 32) {
    v16h a = load_frag(XC + (size_t)row0 * DIq + k, DIq, lane);
#pragma unroll
    for (int t = 0; t < 4; ++t) {
      v16h bf = load_frag(WT + (size_t)(16 * t) * DIq + k, DIq, lane);
      acc[t] = wmma_f16(a, bf, acc[t]);
    }
  }
  const int n = lane & 15, half = lane >> 4;
#pragma unroll
  for (int t = 0; t < 4; ++t)
#pragma unroll
    for (int r = 0; r < 8; ++r) {
      int col = 16 * t + n;
      if (col < XDW)
        XD[(size_t)(row0 + r + 8 * half) * XDW + col] = acc[t][r];
    }
}

// ---------------- 7) dt = softplus(xdbl[:, :24] @ W_dt + dt_bias) ----------------
__global__ void dt_kernel(const float* __restrict__ XD, const float* __restrict__ Wdt,
                          const float* __restrict__ dtb, float* __restrict__ DT) {
  const int row = blockIdx.x;
  __shared__ float xr[Rq];
  if (threadIdx.x < Rq) xr[threadIdx.x] = XD[(size_t)row * XDW + threadIdx.x];
  __syncthreads();
  for (int d = threadIdx.x; d < DIq; d += 256) {
    float acc = dtb[d];
#pragma unroll
    for (int r = 0; r < Rq; ++r) acc += xr[r] * Wdt[r * DIq + d];
    DT[(size_t)row * DIq + d] = (acc > 20.0f) ? acc : log1pf(__expf(acc));
  }
}

// ---------------- 8) selective scan + gating ----------------
__global__ void scan_kernel(const float* __restrict__ XD, const float* __restrict__ DT,
                            const float* __restrict__ XC, const float* __restrict__ XZ,
                            const float* __restrict__ Alog, const float* __restrict__ Dv,
                            _Float16* __restrict__ Y) {
  const int b = blockIdx.x / 3, chunk = blockIdx.x % 3;
  const int d = chunk * 256 + threadIdx.x;
  float A[Sq];
#pragma unroll
  for (int s = 0; s < Sq; ++s) A[s] = -__expf(Alog[d * Sq + s]);
  const float Dd = Dv[d];
  float hs[Sq] = {};
  __shared__ float BC[2 * Sq];
  for (int t = 0; t < Nq; ++t) {
    const int row = (b << 11) + t;
    if (threadIdx.x < 2 * Sq) BC[threadIdx.x] = XD[(size_t)row * XDW + Rq + threadIdx.x];
    __syncthreads();
    const float dt_t = DT[(size_t)row * DIq + d];
    const float x_t  = XC[(size_t)row * DIq + d];
    const float dx   = dt_t * x_t;
    float y = 0.0f;
#pragma unroll
    for (int s = 0; s < Sq; ++s) {
      float dA = __expf(dt_t * A[s]);
      hs[s] = hs[s] * dA + dx * BC[s];
      y += hs[s] * BC[Sq + s];
    }
    y += x_t * Dd;
    const float z = XZ[(size_t)row * XZW + DIq + d];
    y *= z / (1.0f + __expf(-z));
    Y[(size_t)row * DIq + d] = (_Float16)y;
    __syncthreads();
  }
}

// ---------------- 9) GEMM out = y(8192x768) * W_out + residual, scatter ----------------
__global__ void gemm_out_kernel(const _Float16* __restrict__ Y, const _Float16* __restrict__ WT,
                                const float* __restrict__ feat_s, const int* __restrict__ order,
                                float* __restrict__ out) {
  const int lane = threadIdx.x & 31, wave = threadIdx.x >> 5;
  const int row0 = blockIdx.x * 16;
  const int col0 = wave * 48;                // 8 waves * 48 = 384 cols
  v8f acc[3] = {};
  for (int k = 0; k < DIq; k += 32) {
    v16h a = load_frag(Y + (size_t)row0 * DIq + k, DIq, lane);
#pragma unroll
    for (int t = 0; t < 3; ++t) {
      v16h bf = load_frag(WT + (size_t)(col0 + 16 * t) * DIq + k, DIq, lane);
      acc[t] = wmma_f16(a, bf, acc[t]);
    }
  }
  const int n = lane & 15, half = lane >> 4;
#pragma unroll
  for (int t = 0; t < 3; ++t)
#pragma unroll
    for (int r = 0; r < 8; ++r) {
      int row = row0 + r + 8 * half;
      int b   = row >> 11;
      int dst = order[row];                  // scatter: out[b, order[b,t]] = sorted result
      int col = col0 + 16 * t + n;
      out[(size_t)((b << 11) + dst) * DMq + col] = acc[t][r] + feat_s[(size_t)row * DMq + col];
    }
}

// ---------------- host side ----------------
extern "C" void kernel_launch(void* const* d_in, const int* in_sizes, int n_in,
                              void* d_out, int out_size, void* d_ws, size_t ws_size,
                              hipStream_t stream) {
  const float* pos    = (const float*)d_in[0];
  const float* feat   = (const float*)d_in[1];
  const float* ln_g   = (const float*)d_in[2];
  const float* ln_b   = (const float*)d_in[3];
  const float* W_in   = (const float*)d_in[4];
  const float* conv_w = (const float*)d_in[5];
  const float* conv_b = (const float*)d_in[6];
  const float* W_x    = (const float*)d_in[7];
  const float* W_dt   = (const float*)d_in[8];
  const float* dt_b   = (const float*)d_in[9];
  const float* A_log  = (const float*)d_in[10];
  const float* Dv     = (const float*)d_in[11];
  const float* W_out  = (const float*)d_in[12];
  float* out = (float*)d_out;

  char* ws = (char*)d_ws;
  size_t off = 0;
  auto take = [&](size_t bytes) { char* p = ws + off; off = (off + bytes + 255) & ~(size_t)255; return p; };

  unsigned*  codes  = (unsigned*)take(ROWS * 4);
  int*       order  = (int*)take(ROWS * 4);
  int*       inv    = (int*)take(ROWS * 4);
  float*     feat_s = (float*)take((size_t)ROWS * DMq * 4);
  _Float16*  h16    = (_Float16*)take((size_t)ROWS * DMq * 2);
  _Float16*  WinT   = (_Float16*)take((size_t)XZW * DMq * 2);
  _Float16*  WxT    = (_Float16*)take((size_t)XDWP * DIq * 2);
  _Float16*  WoutT  = (_Float16*)take((size_t)DMq * DIq * 2);
  float*     XZ     = (float*)take((size_t)ROWS * XZW * 4);
  float*     xc32   = (float*)take((size_t)ROWS * DIq * 4);
  _Float16*  xc16   = (_Float16*)take((size_t)ROWS * DIq * 2);
  float*     XD     = (float*)take((size_t)ROWS * XDW * 4);
  float*     DT     = (float*)take((size_t)ROWS * DIq * 4);
  _Float16*  Y16    = (_Float16*)take((size_t)ROWS * DIq * 2);
  (void)ws_size; (void)in_sizes; (void)n_in; (void)out_size; (void)inv;

  // weight conversion (f32 -> f16, transposed to N x K)
  wt_f16_kernel<<<(XZW * DMq + 255) / 256, 256, 0, stream>>>(W_in, WinT, DMq, XZW, XZW);
  wt_f16_kernel<<<(XDWP * DIq + 255) / 256, 256, 0, stream>>>(W_x, WxT, DIq, XDW, XDWP);
  wt_f16_kernel<<<(DMq * DIq + 255) / 256, 256, 0, stream>>>(W_out, WoutT, DIq, DMq, DMq);

  morton_kernel<<<Bq, 256, 0, stream>>>(pos, codes);
  sort_kernel<<<Bq, 1024, 0, stream>>>(codes, order, inv);
  gather_ln_kernel<<<ROWS, 128, 0, stream>>>(feat, order, ln_g, ln_b, feat_s, h16);

  gemm_xz_kernel<<<dim3(ROWS / 16, XZW / 512), 256, 0, stream>>>(h16, WinT, XZ);
  conv_silu_kernel<<<(ROWS * DIq + 255) / 256, 256, 0, stream>>>(XZ, conv_w, conv_b, xc32, xc16);
  gemm_xdbl_kernel<<<ROWS / 128, 256, 0, stream>>>(xc16, WxT, XD);
  dt_kernel<<<ROWS, 256, 0, stream>>>(XD, W_dt, dt_b, DT);
  scan_kernel<<<Bq * 3, 256, 0, stream>>>(XD, DT, xc32, XZ, A_log, Dv, Y16);
  gemm_out_kernel<<<ROWS / 16, 256, 0, stream>>>(Y16, WoutT, feat_s, order, out);
}
